// LatentCompressedAttention_88605175316805
// MI455X (gfx1250) — matrix operand order repaired and checked by reading, not compiled, not run on hardware
//
#include <hip/hip_runtime.h>
#include <stdint.h>

typedef __attribute__((ext_vector_type(16))) __bf16 v16bf;
typedef __attribute__((ext_vector_type(8)))  float  v8f;

#define HD     2048      // model dim == H*Dq == H*Dk
#define SEQ    2048
#define BATCH  2
#define NH     16
#define DHEAD  128
#define LAT    512
#define MROWS  (BATCH*SEQ)   // 4096
#define LDSPAD 40            // bf16 elements per LDS row (32 + 8 pad)

#define WMMA_BF16(A,B,C) \
  __builtin_amdgcn_wmma_f32_16x16x32_bf16(false,(A),false,(B),(short)0,(C),false,false)

static __device__ __forceinline__ __bf16 f2bf(float f) {
  unsigned u = __builtin_bit_cast(unsigned, f);
  u += 0x7FFFu + ((u >> 16) & 1u);                 // round-to-nearest-even
  unsigned short h = (unsigned short)(u >> 16);
  return __builtin_bit_cast(__bf16, h);
}
static __device__ __forceinline__ float bf2f(__bf16 b) {
  unsigned short h = __builtin_bit_cast(unsigned short, b);
  unsigned u = ((unsigned)h) << 16;
  return __builtin_bit_cast(float, u);
}

union FragBF { uint4 u[2]; v16bf v; };

// A-operand fragment (16x32, MxK): the lane's 16 elements are
// K = hi*16 + (0..15), contiguous.  p must already include the
// lane's (row, +hi*16) offset; loads 32 contiguous bytes.
static __device__ __forceinline__ v16bf load_frag_a(const __bf16* p) {
  FragBF f;
  const uint4* q = (const uint4*)p;
  f.u[0] = q[0];
  f.u[1] = q[1];   // next 8 elements, contiguous
  return f.v;
}

// B-operand fragment (32x16, KxN): in this code lanes 0-15 hold K=0..15 and lanes
// 16-31 hold K=16..31, contiguous within the lane.  p must already include the
// lane's (row=n, +hi*16) offset; loads 32 contiguous bytes.
static __device__ __forceinline__ v16bf load_frag_b(const __bf16* p) {
  FragBF f;
  const uint4* q = (const uint4*)p;
  f.u[0] = q[0];
  f.u[1] = q[1];   // next 8 elements, contiguous
  return f.v;
}

// ---------------------------------------------------------------- conversions
__global__ void k_f32_to_bf16(const float* __restrict__ src,
                              __bf16* __restrict__ dst, int n) {
  int i = blockIdx.x * 256 + threadIdx.x;
  if (i < n) dst[i] = f2bf(src[i]);
}

// src is K x N row-major f32; dst is N x K row-major bf16 (i.e. src^T)
__global__ void k_transpose_bf16(const float* __restrict__ src,
                                 __bf16* __restrict__ dst, int K, int N) {
  int i = blockIdx.x * 256 + threadIdx.x;
  if (i < K * N) {
    int k = i / N, n = i - k * N;
    dst[(size_t)n * K + k] = f2bf(src[i]);
  }
}

// ---------------------------------------------------------------- RoPE + scale
// Q is [MROWS][HD] bf16, cols h*128+d. Rotates dims [0,64), scales all dims.
// Each thread owns the elements it both reads and writes (pair-safe in-place).
__global__ void k_rope_scale(__bf16* __restrict__ q,
                             const int* __restrict__ positions) {
  int gid = blockIdx.x * 256 + threadIdx.x;
  int j   = gid & 63;
  int h   = (gid >> 6) & 15;
  int row = gid >> 10;                 // b*SEQ + s
  if (row >= MROWS) return;
  const float SCALE = 0.08838834764831845f;   // 1/sqrt(128)
  size_t base = (size_t)row * HD + h * DHEAD;
  if (j < 32) {
    float pos = (float)positions[row & (SEQ - 1)];
    float inv = __expf(-((2.0f * (float)j) / 64.0f) * 9.210340371976184f);
    float fr  = pos * inv;
    float s, c;
    __sincosf(fr, &s, &c);
    float x1 = bf2f(q[base + j]);
    float x2 = bf2f(q[base + j + 32]);
    q[base + j]      = f2bf((x1 * c - x2 * s) * SCALE);
    q[base + j + 32] = f2bf((x1 * s + x2 * c) * SCALE);
  } else {
    int d = j + 32;                    // covers dims 64..95 and 96..127
    q[base + d]      = f2bf(bf2f(q[base + d]) * SCALE);
    q[base + d + 32] = f2bf(bf2f(q[base + d + 32]) * SCALE);
  }
}

// ---------------------------------------------------------------- GEMM
// C[M,N] = A[M,K] @ B[K,N], with B supplied TRANSPOSED (Bt[N][K], ld = K).
// Block tile 128x128, 8 waves (wave32), per-wave 2(M)x4(N) 16x16 WMMA tiles.
// outMode: 0 = bf16 row-major, 1 = bf16 V-transposed [b,h,d,s], 2 = f32 row-major
__global__ void __launch_bounds__(256) k_gemm_bf16(
    const __bf16* __restrict__ A, const __bf16* __restrict__ Bt,
    void* __restrict__ Cout, int N, int K, int outMode) {
  __shared__ __bf16 sA[128 * LDSPAD];
  __shared__ __bf16 sB[128 * LDSPAD];
  const int tid  = threadIdx.x;
  const int lane = tid & 31, wid = tid >> 5;
  const int m16  = lane & 15, hi = lane >> 4;
  const int wm   = wid & 3,   wn = wid >> 2;
  const int bm   = blockIdx.y * 128, bn = blockIdx.x * 128;

  v8f acc[2][4];
  const v8f vzero = {0.f,0.f,0.f,0.f,0.f,0.f,0.f,0.f};
  #pragma unroll
  for (int i = 0; i < 2; i++)
    #pragma unroll
    for (int j = 0; j < 4; j++) acc[i][j] = vzero;

  for (int kk = 0; kk < K; kk += 32) {
    if (kk) __syncthreads();           // all waves done reading previous tiles
    // Stage A (128x32) and Bt (128x32) tiles: 512 x 16B chunks each,
    // 2 chunks per thread per tile, via CDNA5 async global->LDS DMA.
    #pragma unroll
    for (int i = 0; i < 2; i++) {
      int chunk = tid + i * 256;
      int row = chunk >> 2, cc = chunk & 3;        // 4 x 16B chunks per row
      unsigned long long ga =
          (unsigned long long)(uintptr_t)(A + (size_t)(bm + row) * K + kk + cc * 8);
      unsigned la = (unsigned)(uintptr_t)&sA[row * LDSPAD + cc * 8];
      asm volatile("global_load_async_to_lds_b128 %0, %1, off"
                   :: "v"(la), "v"(ga) : "memory");
      unsigned long long gb =
          (unsigned long long)(uintptr_t)(Bt + (size_t)(bn + row) * K + kk + cc * 8);
      unsigned lb = (unsigned)(uintptr_t)&sB[row * LDSPAD + cc * 8];
      asm volatile("global_load_async_to_lds_b128 %0, %1, off"
                   :: "v"(lb), "v"(gb) : "memory");
    }
    asm volatile("s_wait_asynccnt 0" ::: "memory");
    __syncthreads();

    v16bf af[2], bfr[4];
    #pragma unroll
    for (int ms = 0; ms < 2; ms++)
      af[ms] = load_frag_a(&sA[(wm * 32 + ms * 16 + m16) * LDSPAD + hi * 16]);
    #pragma unroll
    for (int ns = 0; ns < 4; ns++)
      bfr[ns] = load_frag_b(&sB[(wn * 64 + ns * 16 + m16) * LDSPAD + hi * 16]);
    #pragma unroll
    for (int ms = 0; ms < 2; ms++)
      #pragma unroll
      for (int ns = 0; ns < 4; ns++)
        acc[ms][ns] = WMMA_BF16(af[ms], bfr[ns], acc[ms][ns]);
  }

  #pragma unroll
  for (int ms = 0; ms < 2; ms++)
    #pragma unroll
    for (int ns = 0; ns < 4; ns++)
      #pragma unroll
      for (int r = 0; r < 8; r++) {
        int row = bm + wm * 32 + ms * 16 + r + 8 * hi;
        int col = bn + wn * 64 + ns * 16 + m16;
        float v = acc[ms][ns][r];
        if (outMode == 2) {
          ((float*)Cout)[(size_t)row * N + col] = v;
        } else if (outMode == 1) {     // V^T layout: [b][h][d][s]
          int b = row >> 11, s = row & 2047;
          int h = col >> 7,  d = col & 127;
          ((__bf16*)Cout)[(((size_t)(b * NH + h) * DHEAD + d) << 11) + s] = f2bf(v);
        } else {
          ((__bf16*)Cout)[(size_t)row * N + col] = f2bf(v);
        }
      }
}

// ---------------------------------------------------------------- attention
// One wave handles 16 queries for one (b,h); flash softmax over SEQ keys in
// 32-key steps. Q pre-scaled by 1/sqrt(128). Vt is [b][h][d][s].
__global__ void __launch_bounds__(256) k_attention(
    const __bf16* __restrict__ Q, const __bf16* __restrict__ Kmat,
    const __bf16* __restrict__ Vt, __bf16* __restrict__ AO) {
  __shared__ __bf16 sP[8][16 * LDSPAD];
  const int tid  = threadIdx.x;
  const int lane = tid & 31, wid = tid >> 5;
  const int m16  = lane & 15, hi = lane >> 4;

  int wtask = blockIdx.x * 8 + wid;
  int qt = wtask & 127;
  int bh = wtask >> 7;
  int h = bh & 15, b = bh >> 4;

  // A-operand base (Q): includes +hi*16 for the contiguous A layout.
  const __bf16* qbase = Q    + ((size_t)(b * SEQ + qt * 16 + m16)) * HD + h * DHEAD + hi * 16;
  // B-operand bases (K rows as Bt[n=key][k=d], Vt rows as Bt[n=d][k=key]):
  // include +hi*16 for the contiguous B layout.
  const __bf16* kbase = Kmat + ((size_t)(b * SEQ)) * HD + h * DHEAD + hi * 16;
  const __bf16* vbase = Vt   + ((size_t)(b * NH + h) * DHEAD) * SEQ + hi * 16;

  v16bf aq[4];
  #pragma unroll
  for (int c = 0; c < 4; c++) aq[c] = load_frag_a(qbase + c * 32);

  const v8f vzero = {0.f,0.f,0.f,0.f,0.f,0.f,0.f,0.f};
  v8f o[8];
  float mi[8], li[8];
  #pragma unroll
  for (int t = 0; t < 8; t++) o[t] = vzero;
  #pragma unroll
  for (int r = 0; r < 8; r++) { mi[r] = -1e30f; li[r] = 0.f; }

  for (int kb = 0; kb < SEQ; kb += 32) {
    v8f s0 = vzero, s1 = vzero;
    #pragma unroll
    for (int c = 0; c < 4; c++) {
      v16bf k0 = load_frag_b(kbase + (size_t)(kb + m16) * HD + c * 32);
      v16bf k1 = load_frag_b(kbase + (size_t)(kb + 16 + m16) * HD + c * 32);
      s0 = WMMA_BF16(aq[c], k0, s0);
      s1 = WMMA_BF16(aq[c], k1, s1);
    }

    float alpha[8];
    #pragma unroll
    for (int r = 0; r < 8; r++) {
      float v0 = s0[r], v1 = s1[r];
      float mx = fmaxf(v0, v1);
      #pragma unroll
      for (int msk = 1; msk < 16; msk <<= 1)
        mx = fmaxf(mx, __shfl_xor(mx, msk, 32));   // stays within 16-lane half
      float mnew = fmaxf(mi[r], mx);
      alpha[r] = __expf(mi[r] - mnew);
      float p0 = __expf(v0 - mnew), p1 = __expf(v1 - mnew);
      float rs = p0 + p1;
      #pragma unroll
      for (int msk = 1; msk < 16; msk <<= 1)
        rs += __shfl_xor(rs, msk, 32);
      li[r] = li[r] * alpha[r] + rs;
      mi[r] = mnew;
      // C-layout -> LDS (row-major 16x32, padded)
      sP[wid][(r + 8 * hi) * LDSPAD + m16]      = f2bf(p0);
      sP[wid][(r + 8 * hi) * LDSPAD + 16 + m16] = f2bf(p1);
    }
    asm volatile("s_wait_dscnt 0" ::: "memory");
    v16bf ap = load_frag_a(&sP[wid][m16 * LDSPAD + hi * 16]);

    #pragma unroll
    for (int t = 0; t < 8; t++)
      #pragma unroll
      for (int r = 0; r < 8; r++) o[t][r] = o[t][r] * alpha[r];

    #pragma unroll
    for (int t = 0; t < 8; t++) {
      v16bf bv = load_frag_b(vbase + (size_t)(t * 16 + m16) * SEQ + kb);
      o[t] = WMMA_BF16(ap, bv, o[t]);
    }
  }

  #pragma unroll
  for (int r = 0; r < 8; r++) {
    float rl = 1.0f / li[r];
    int row = b * SEQ + qt * 16 + r + 8 * hi;
    #pragma unroll
    for (int t = 0; t < 8; t++) {
      int col = h * DHEAD + t * 16 + m16;
      AO[(size_t)row * HD + col] = f2bf(o[t][r] * rl);
    }
  }
}

// ---------------------------------------------------------------- launch
extern "C" void kernel_launch(void* const* d_in, const int* in_sizes, int n_in,
                              void* d_out, int out_size, void* d_ws, size_t ws_size,
                              hipStream_t stream) {
  (void)in_sizes; (void)n_in; (void)out_size; (void)ws_size;
  const float* x  = (const float*)d_in[0];
  const int*   pp = (const int*)d_in[1];
  const float* Wq = (const float*)d_in[2];
  const float* Wc = (const float*)d_in[3];
  const float* Wk = (const float*)d_in[4];
  const float* Wv = (const float*)d_in[5];
  const float* Wo = (const float*)d_in[6];

  __bf16* ws = (__bf16*)d_ws;
  size_t off = 0;
  __bf16* XB  = ws + off; off += (size_t)MROWS * HD;   // x in bf16
  __bf16* WQT = ws + off; off += (size_t)HD * HD;      // Wq^T
  __bf16* WCT = ws + off; off += (size_t)LAT * HD;     // Wc^T
  __bf16* WKT = ws + off; off += (size_t)HD * LAT;     // Wk^T
  __bf16* WVT = ws + off; off += (size_t)HD * LAT;     // Wv^T
  __bf16* WOT = ws + off; off += (size_t)HD * HD;      // Wo^T
  __bf16* QB  = ws + off; off += (size_t)MROWS * HD;   // Q (roped, scaled)
  __bf16* CB  = ws + off; off += (size_t)MROWS * LAT;  // latent
  __bf16* KB  = ws + off; off += (size_t)MROWS * HD;   // K
  __bf16* VT  = ws + off; off += (size_t)MROWS * HD;   // V transposed
  __bf16* AO  = ws + off; off += (size_t)MROWS * HD;   // attention out

  dim3 blk(256);
  k_f32_to_bf16 <<<(MROWS * HD) / 256, blk, 0, stream>>>(x, XB, MROWS * HD);
  k_transpose_bf16<<<(HD * HD) / 256,  blk, 0, stream>>>(Wq, WQT, HD, HD);
  k_transpose_bf16<<<(HD * LAT) / 256, blk, 0, stream>>>(Wc, WCT, HD, LAT);
  k_transpose_bf16<<<(LAT * HD) / 256, blk, 0, stream>>>(Wk, WKT, LAT, HD);
  k_transpose_bf16<<<(LAT * HD) / 256, blk, 0, stream>>>(Wv, WVT, LAT, HD);
  k_transpose_bf16<<<(HD * HD) / 256,  blk, 0, stream>>>(Wo, WOT, HD, HD);

  k_gemm_bf16<<<dim3(HD / 128,  MROWS / 128), blk, 0, stream>>>(XB, WQT, QB, HD,  HD,  0);
  k_gemm_bf16<<<dim3(LAT / 128, MROWS / 128), blk, 0, stream>>>(XB, WCT, CB, LAT, HD,  0);
  k_rope_scale<<<(MROWS * NH * 64) / 256, blk, 0, stream>>>(QB, pp);
  k_gemm_bf16<<<dim3(HD / 128, MROWS / 128), blk, 0, stream>>>(CB, WKT, KB, HD, LAT, 0);
  k_gemm_bf16<<<dim3(HD / 128, MROWS / 128), blk, 0, stream>>>(CB, WVT, VT, HD, LAT, 1);

  k_attention<<<512, blk, 0, stream>>>(QB, KB, VT, AO);

  k_gemm_bf16<<<dim3(HD / 128, MROWS / 128), blk, 0, stream>>>(AO, WOT, d_out, HD, HD, 2);
}
